// InfoNCELoss_3427383902249
// MI455X (gfx1250) — matrix operand order repaired and checked by reading, not compiled, not run on hardware
//
#include <hip/hip_runtime.h>
#include <hip/hip_bf16.h>
#include <math.h>

typedef __attribute__((ext_vector_type(16))) _Float16 v16h;
typedef __attribute__((ext_vector_type(8)))  _Float16 v8h;
typedef __attribute__((ext_vector_type(8)))  float    v8f;

#define DDIM   768
#define NROWS  8192
#define BHALF  4096
#define EPS    1e-8f
// sim = cos * (1/T), 1/T = 2; bounded in [-2,2] -> fixed lse max = 2.0
// s += exp(sim - 2) = exp2(cos * (2*log2e) + (-2*log2e))
#define K1  2.885390081777927f
#define K0 -2.885390081777927f
#define PANEL    64
#define NPANELS  (NROWS / PANEL)   // 128
#define NSPLITS  4                 // blocks per strip-pair (column-sweep split)

// raw v_exp_f32: argument is bounded in [-5.78, 0] -> no subnormal fixup needed
#if __has_builtin(__builtin_amdgcn_exp2f)
#define FAST_EXP2(x) __builtin_amdgcn_exp2f(x)
#else
#define FAST_EXP2(x) exp2f(x)
#endif

// ---------------------------------------------------------------------------
// Kernel 1: row-wise L2 normalization, f32 -> f16 (hn written to workspace)
// ---------------------------------------------------------------------------
__launch_bounds__(256)
__global__ void normalize_kernel(const float* __restrict__ a1,
                                 const float* __restrict__ a2,
                                 _Float16* __restrict__ hn) {
    const int row = blockIdx.x;
    const float* src = (row < BHALF) ? (a1 + (size_t)row * DDIM)
                                     : (a2 + (size_t)(row - BHALF) * DDIM);
    float ss = 0.0f;
    for (int i = threadIdx.x; i < DDIM; i += 256) {
        float v = src[i];
        ss += v * v;
    }
    __shared__ float red[8];
    #pragma unroll
    for (int off = 16; off > 0; off >>= 1)
        ss += __shfl_down(ss, off, 32);
    if ((threadIdx.x & 31) == 0) red[threadIdx.x >> 5] = ss;
    __syncthreads();
    if (threadIdx.x < 8) {
        float v = red[threadIdx.x];
        #pragma unroll
        for (int off = 4; off > 0; off >>= 1)
            v += __shfl_down(v, off, 32);
        if (threadIdx.x == 0) red[0] = v;
    }
    __syncthreads();
    const float inv = 1.0f / fmaxf(sqrtf(red[0]), EPS);
    for (int i = threadIdx.x; i < DDIM; i += 256)
        hn[(size_t)row * DDIM + i] = (_Float16)(src[i] * inv);
}

// ---------------------------------------------------------------------------
// Kernel 2: zero rowsum accumulators + scalar output (graph-capture safe)
// ---------------------------------------------------------------------------
__global__ void zero_kernel(float* __restrict__ rowsum, float* __restrict__ out) {
    const int i = blockIdx.x * 256 + threadIdx.x;
    if (i < NROWS) rowsum[i] = 0.0f;
    if (i == 0) out[0] = 0.0f;
}

// ---------------------------------------------------------------------------
// Kernel 3: SYMMETRIC fused sim-GEMM, upper-triangular 64x64 panels only.
// Each element feeds row i (register sums) and row j (column-sum atomics).
// Strip-pair p = blockIdx/NSPLITS handles I=p and I=127-p (129 panels,
// balanced); the NSPLITS blocks of a pair split the column sweep 16 ways
// (cwG = split*4 + cw, J = I + cwG step 16). 8 waves = mw(2) x cw(4);
// 2x4 WMMA register tile per wave (8 wmma / k-step, 1.5 b128-loads/wmma).
// ---------------------------------------------------------------------------
__launch_bounds__(256, 1)
__global__ void infonce_panels_kernel(const _Float16* __restrict__ hn,
                                      float* __restrict__ rowsum,
                                      float* __restrict__ posArr) {
    __shared__ _Float16 As[PANEL][DDIM];   // 96 KB (CDNA5: 320 KB/WGP)

    const int lane  = threadIdx.x & 31;
    const int wave  = threadIdx.x >> 5;
    const int mw    = wave >> 2;             // M half: rows mw*32..mw*32+31
    const int cw    = wave & 3;
    const int split = blockIdx.x & (NSPLITS - 1);
    const int pair  = blockIdx.x / NSPLITS;
    const int cwG   = split * 4 + cw;        // 0..15 column-sweep phase

    const int laneLo = lane & 15;
    const int laneHi = lane >> 4;
    const int rowLaneOff = laneHi ? 8 : 0;
    bool dsel[8];
    #pragma unroll
    for (int r = 0; r < 8; ++r) dsel[r] = (laneLo == r + rowLaneOff);

    const _Float16* aPtr0 = &As[mw * 32 + laneLo][laneHi * 8];
    const _Float16* aPtr1 = &As[mw * 32 + 16 + laneLo][laneHi * 8];

    for (int p = 0; p < 2; ++p) {
        const int I = p ? (NPANELS - 1 - pair) : pair;
        const int rowBase = I * PANEL;

        // cooperative A-panel copy (rows contiguous in hn): 64x768 halves
        {
            const float4* srcv = (const float4*)(hn + (size_t)rowBase * DDIM);
            float4* dstv = (float4*)&As[0][0];
            for (int i = threadIdx.x; i < PANEL * DDIM / 8; i += blockDim.x)
                dstv[i] = srcv[i];
        }
        __syncthreads();

        float srun[2][8];
        #pragma unroll
        for (int m = 0; m < 2; ++m)
            #pragma unroll
            for (int r = 0; r < 8; ++r) srun[m][r] = 0.0f;

        for (int J = I + cwG; J < NPANELS; J += 4 * NSPLITS) {
            const int colBase = J * PANEL;
            const _Float16* bBase =
                hn + (size_t)(colBase + laneLo) * DDIM + laneHi * 8;

            v8f cc[2][4];
            #pragma unroll
            for (int m = 0; m < 2; ++m)
                #pragma unroll
                for (int n = 0; n < 4; ++n) cc[m][n] = (v8f){};

            #pragma unroll 2
            for (int k0 = 0; k0 < DDIM; k0 += 32) {
                v16h a[2];
                {
                    v8h l0 = *(const v8h*)(aPtr0 + k0);
                    v8h h0 = *(const v8h*)(aPtr0 + k0 + 16);
                    a[0] = __builtin_shufflevector(l0, h0,
                            0,1,2,3,4,5,6,7,8,9,10,11,12,13,14,15);
                    v8h l1 = *(const v8h*)(aPtr1 + k0);
                    v8h h1 = *(const v8h*)(aPtr1 + k0 + 16);
                    a[1] = __builtin_shufflevector(l1, h1,
                            0,1,2,3,4,5,6,7,8,9,10,11,12,13,14,15);
                }
                v16h b[4];
                #pragma unroll
                for (int n = 0; n < 4; ++n)
                    b[n] = __builtin_shufflevector(
                            *(const v8h*)(bBase + (size_t)n * 16 * DDIM + k0),
                            *(const v8h*)(bBase + (size_t)n * 16 * DDIM + k0 + 16),
                            0,1,2,3,4,5,6,7,8,9,10,11,12,13,14,15);
                #pragma unroll
                for (int m = 0; m < 2; ++m)
                    #pragma unroll
                    for (int n = 0; n < 4; ++n)
                        cc[m][n] = __builtin_amdgcn_wmma_f32_16x16x32_f16(
                            false, a[m], false, b[n], (short)0, cc[m][n], false, false);
            }

            if (J == I) {
                // diagonal panel: row sums only; mask the true diagonal
                // (N-tile n == 2*mw + m, slot dsel[r]); no column sums
                #pragma unroll
                for (int m = 0; m < 2; ++m)
                    #pragma unroll
                    for (int n = 0; n < 4; ++n)
                        #pragma unroll
                        for (int r = 0; r < 8; ++r) {
                            float e = FAST_EXP2(fmaf(cc[m][n][r], K1, K0));
                            if (n == 2 * mw + m && dsel[r]) e = 0.0f;
                            srun[m][r] += e;
                        }
            } else if (J == I + BHALF / PANEL) {
                // positive panel: panel-local diagonal = the positive pair
                float colsum[4] = {0.0f, 0.0f, 0.0f, 0.0f};
                #pragma unroll
                for (int m = 0; m < 2; ++m)
                    #pragma unroll
                    for (int n = 0; n < 4; ++n)
                        #pragma unroll
                        for (int r = 0; r < 8; ++r) {
                            float e = FAST_EXP2(fmaf(cc[m][n][r], K1, K0));
                            srun[m][r] += e;
                            colsum[n] += e;
                            if (n == 2 * mw + m && dsel[r]) {
                                const float v = cc[m][n][r] * 2.0f;
                                const int gRow = rowBase + mw * 32 + m * 16 + r + rowLaneOff;
                                posArr[gRow] = v;
                                posArr[gRow + BHALF] = v;   // col = row + B
                            }
                        }
                #pragma unroll
                for (int n = 0; n < 4; ++n) {
                    colsum[n] += __shfl_xor(colsum[n], 16, 32);
                    if (laneHi == 0)
                        atomicAdd(&rowsum[colBase + n * 16 + laneLo], colsum[n]);
                }
            } else {
                // hot path: pure fma / v_exp / add, no per-element conditionals
                float colsum[4] = {0.0f, 0.0f, 0.0f, 0.0f};
                #pragma unroll
                for (int m = 0; m < 2; ++m)
                    #pragma unroll
                    for (int n = 0; n < 4; ++n)
                        #pragma unroll
                        for (int r = 0; r < 8; ++r) {
                            float e = FAST_EXP2(fmaf(cc[m][n][r], K1, K0));
                            srun[m][r] += e;
                            colsum[n] += e;
                        }
                #pragma unroll
                for (int n = 0; n < 4; ++n) {
                    colsum[n] += __shfl_xor(colsum[n], 16, 32);
                    if (laneHi == 0)
                        atomicAdd(&rowsum[colBase + n * 16 + laneLo], colsum[n]);
                }
            }
        }

        // flush register row sums: reduce across 16-lane halves, then atomics
        #pragma unroll
        for (int m = 0; m < 2; ++m)
            #pragma unroll
            for (int r = 0; r < 8; ++r) {
                #pragma unroll
                for (int off = 8; off > 0; off >>= 1)
                    srun[m][r] += __shfl_xor(srun[m][r], off, 32);
            }
        if (laneLo == 0) {
            #pragma unroll
            for (int m = 0; m < 2; ++m)
                #pragma unroll
                for (int r = 0; r < 8; ++r)
                    atomicAdd(&rowsum[rowBase + mw * 32 + m * 16 + r + rowLaneOff],
                              srun[m][r]);
        }
        __syncthreads();   // protect As before next phase overwrites it
    }
}

// ---------------------------------------------------------------------------
// Kernel 4: loss = mean over rows of (2 + ln(rowsum) - pos)
// ---------------------------------------------------------------------------
__launch_bounds__(256)
__global__ void finish_kernel(const float* __restrict__ rowsum,
                              const float* __restrict__ posArr,
                              float* __restrict__ out) {
    const int i = blockIdx.x * 256 + threadIdx.x;
    float c = 2.0f + logf(rowsum[i]) - posArr[i];
    #pragma unroll
    for (int off = 16; off > 0; off >>= 1)
        c += __shfl_xor(c, off, 32);
    __shared__ float red[8];
    if ((threadIdx.x & 31) == 0) red[threadIdx.x >> 5] = c;
    __syncthreads();
    if (threadIdx.x == 0) {
        float s = 0.0f;
        #pragma unroll
        for (int w = 0; w < 8; ++w) s += red[w];
        atomicAdd(out, s * (1.0f / (float)NROWS));
    }
}

// ---------------------------------------------------------------------------
extern "C" void kernel_launch(void* const* d_in, const int* in_sizes, int n_in,
                              void* d_out, int out_size, void* d_ws, size_t ws_size,
                              hipStream_t stream) {
    const float* a1 = (const float*)d_in[0];
    const float* a2 = (const float*)d_in[1];
    float* out = (float*)d_out;

    _Float16* hn   = (_Float16*)d_ws;                       // 12 MB
    float* rowsum  = (float*)((char*)d_ws + (size_t)NROWS * DDIM * 2);
    float* posArr  = rowsum + NROWS;

    normalize_kernel<<<NROWS, 256, 0, stream>>>(a1, a2, hn);
    zero_kernel<<<NROWS / 256, 256, 0, stream>>>(rowsum, out);
    infonce_panels_kernel<<<(NPANELS / 2) * NSPLITS, 256, 0, stream>>>(hn, rowsum, posArr);
    finish_kernel<<<NROWS / 256, 256, 0, stream>>>(rowsum, posArr, out);
}
